// MixtralSparseMoeBlock_46170898432109
// MI455X (gfx1250) — matrix operand order repaired and checked by reading, not compiled, not run on hardware
//
#include <hip/hip_runtime.h>
#include <hip/hip_bf16.h>
#include <math.h>

typedef __attribute__((ext_vector_type(16))) __bf16 v16bf;
typedef __attribute__((ext_vector_type(8)))  __bf16 v8bf;
typedef __attribute__((ext_vector_type(8)))  float  v8f;

#define HIDDEN 1024
#define FFN    3584
#define NEXP   8

// ---------------------------------------------------------------------------
// Router: one wave (32 lanes) per token. Computes logits = x @ gate_w,
// softmax, top-2, normalized combine weights. Logits go to d_out tail.
// ---------------------------------------------------------------------------
__global__ void moe_router(const float* __restrict__ x,
                           const float* __restrict__ gw,
                           float* __restrict__ logits_out,
                           int* __restrict__ eA, int* __restrict__ eB,
                           float* __restrict__ wAo, float* __restrict__ wBo,
                           int T)
{
    int wave = threadIdx.x >> 5;
    int lane = threadIdx.x & 31;
    int t = blockIdx.x * 8 + wave;
    if (t >= T) return;

    float acc[NEXP];
#pragma unroll
    for (int e = 0; e < NEXP; ++e) acc[e] = 0.f;

    const float* xr = x + (size_t)t * HIDDEN;
    for (int c = lane; c < HIDDEN; c += 32) {
        float xv = xr[c];
        const float* g = gw + (size_t)c * NEXP;
#pragma unroll
        for (int e = 0; e < NEXP; ++e) acc[e] += xv * g[e];
    }
#pragma unroll
    for (int e = 0; e < NEXP; ++e)
        for (int off = 16; off > 0; off >>= 1)
            acc[e] += __shfl_xor(acc[e], off, 32);

    if (lane == 0) {
        float m = acc[0];
        for (int e = 1; e < NEXP; ++e) m = fmaxf(m, acc[e]);
        float p[NEXP];
        for (int e = 0; e < NEXP; ++e) p[e] = expf(acc[e] - m);
        // top-2 (first index wins ties, matching lax.top_k)
        int a = 0; float pa = p[0];
        for (int e = 1; e < NEXP; ++e) if (p[e] > pa) { pa = p[e]; a = e; }
        int b = (a == 0) ? 1 : 0; float pb = p[b];
        for (int e = 0; e < NEXP; ++e)
            if (e != a && p[e] > pb) { pb = p[e]; b = e; }
        float inv = 1.f / (pa + pb);
        eA[t] = a; eB[t] = b; wAo[t] = pa * inv; wBo[t] = pb * inv;
        for (int e = 0; e < NEXP; ++e)
            logits_out[(size_t)t * NEXP + e] = acc[e];
    }
}

__global__ void moe_zero_counts(int* c)
{
    if (threadIdx.x < NEXP) c[threadIdx.x] = 0;
}

// Counting-sort tokens into per-expert lists; entry = token*2 + rank.
__global__ void moe_scatter(const int* __restrict__ eA, const int* __restrict__ eB,
                            int* __restrict__ counts, int* __restrict__ lists, int T)
{
    int t = blockIdx.x * 256 + threadIdx.x;
    if (t >= T) return;
    int a = eA[t];
    int sa = atomicAdd(&counts[a], 1);
    lists[(size_t)a * T + sa] = t * 2;
    int b = eB[t];
    int sb = atomicAdd(&counts[b], 1);
    lists[(size_t)b * T + sb] = t * 2 + 1;
}

// ---------------------------------------------------------------------------
// Pack fp32 weights [E][K][N] into bf16 WMMA B-fragment tiles:
// tile (e, nt, kt) = 32x16 (KxN), stored as 32 lanes x 16 contiguous bf16:
//   lane l holds column N = nt*16 + (l&15), K = kt*32 + (l&16?8:0) + (j<8 ? j : j+8).
// ---------------------------------------------------------------------------
__global__ void moe_pack_bf16(const float* __restrict__ src,
                              __bf16* __restrict__ dst,
                              int E, int K, int N)
{
    size_t gid = (size_t)blockIdx.x * 256 + threadIdx.x;
    int lane = (int)(gid & 31);
    size_t tile = gid >> 5;
    int Kt = K / 32, Nt = N / 16;
    size_t tilesPerE = (size_t)Kt * Nt;
    if (tile >= (size_t)E * tilesPerE) return;
    int e = (int)(tile / tilesPerE);
    size_t r = tile % tilesPerE;
    int nt = (int)(r / Kt);
    int kt = (int)(r % Kt);
    int nn = nt * 16 + (lane & 15);
    int kb = kt * 32 + ((lane & 16) ? 8 : 0);

    union { v16bf v; __bf16 e16[16]; } u;
    const float* s = src + ((size_t)e * K + kb) * N + nn;
#pragma unroll
    for (int j = 0; j < 16; ++j) u.e16[j] = (__bf16)s[(size_t)(j < 8 ? j : j + 8) * N];
    *(v16bf*)(dst + tile * 512 + (size_t)lane * 16) = u.v;
}

// ---------------------------------------------------------------------------
// Fused expert MLP: block = one 16-token tile of one expert, 8 waves.
// contrib[rank][token][H] written exclusively (deterministic, no fp atomics).
// ---------------------------------------------------------------------------
struct __align__(32) MoeSmem {
    __bf16 xT[16 * HIDDEN];   // 32 KB: token tile, bf16
    __bf16 hT[16 * 128];      // 4 KB : silu(g)*u*weight chunk, bf16
    int    tokrk[16];         // token*2 + rank per row
    float  w[16];             // combine weight per row
};

__global__ void __launch_bounds__(256)
moe_expert_gemm(const float* __restrict__ x,
                const __bf16* __restrict__ w1p,
                const __bf16* __restrict__ w3p,
                const __bf16* __restrict__ w2p,
                const int* __restrict__ counts,
                const int* __restrict__ lists,
                const float* __restrict__ wA, const float* __restrict__ wB,
                float* __restrict__ c0, float* __restrict__ c1, int T)
{
    __shared__ MoeSmem sm;
    int e = blockIdx.y;
    int cnt = counts[e];
    int row0 = blockIdx.x * 16;
    if (row0 >= cnt) return;           // uniform early-out (no barriers yet)
    int rowsLeft = cnt - row0;

    int tid = threadIdx.x;
    int wave = tid >> 5, lane = tid & 31;
    int mlo = lane & 15;
    int koffA = (lane & 16) ? 8 : 0;   // 16-bit A-matrix lane K offset

    if (tid < 16) {
        int trk = 0; float wgt = 0.f;
        if (tid < rowsLeft) {
            trk = lists[(size_t)e * T + row0 + tid];
            int tok = trk >> 1;
            wgt = (trk & 1) ? wB[tok] : wA[tok];
        }
        sm.tokrk[tid] = trk;
        sm.w[tid] = wgt;
    }
    __syncthreads();

    // Stage x tile as bf16 (coalesced along H).
    for (int idx = tid; idx < 16 * HIDDEN; idx += 256) {
        int rr = idx >> 10, cc = idx & (HIDDEN - 1);
        int tok = sm.tokrk[rr] >> 1;
        sm.xT[idx] = (__bf16)x[(size_t)tok * HIDDEN + cc];
    }
    __syncthreads();

    v8f acc[8] = {};                   // 16x128 H-slice accumulators per wave

    for (int fc = 0; fc < FFN / 128; ++fc) {
        // ---- phase 1: g = x@w1, u = x@w3 for this wave's 16x16 F-tile ----
        v8f g = {};
        v8f uacc = {};
        int nt = fc * 8 + wave;
        const __bf16* b1base =
            w1p + ((size_t)(e * (FFN / 16) + nt) * (HIDDEN / 32)) * 512 + (size_t)lane * 16;
        const __bf16* b3base =
            w3p + ((size_t)(e * (FFN / 16) + nt) * (HIDDEN / 32)) * 512 + (size_t)lane * 16;
        for (int kt = 0; kt < HIDDEN / 32; ++kt) {
            union { v16bf v; v8bf h[2]; } a;
            const __bf16* ap = &sm.xT[mlo * HIDDEN + kt * 32 + koffA];
            a.h[0] = *(const v8bf*)ap;          // K = koff .. koff+7
            a.h[1] = *(const v8bf*)(ap + 16);   // K = koff+16 .. koff+23
            v16bf b1 = *(const v16bf*)(b1base + (size_t)kt * 512);
            v16bf b3 = *(const v16bf*)(b3base + (size_t)kt * 512);
            g    = __builtin_amdgcn_wmma_f32_16x16x32_bf16(false, a.v, false, b1,
                                                           (short)0, g,    false, false);
            uacc = __builtin_amdgcn_wmma_f32_16x16x32_bf16(false, a.v, false, b3,
                                                           (short)0, uacc, false, false);
        }
        // ---- silu(g)*u * combine_weight -> bf16 h chunk in LDS ----
#pragma unroll
        for (int r = 0; r < 8; ++r) {
            int m = r + ((lane & 16) ? 8 : 0);
            float gg = g[r];
            float hv = (gg / (1.f + expf(-gg))) * uacc[r] * sm.w[m];
            sm.hT[m * 128 + wave * 16 + mlo] = (__bf16)hv;
        }
        __syncthreads();
        // ---- phase 2: down-proj, each wave owns H cols [wave*128, +128) ----
#pragma unroll
        for (int k2 = 0; k2 < 4; ++k2) {
            union { v16bf v; v8bf h[2]; } a;
            const __bf16* ap = &sm.hT[mlo * 128 + k2 * 32 + koffA];
            a.h[0] = *(const v8bf*)ap;
            a.h[1] = *(const v8bf*)(ap + 16);
            int ktg = fc * 4 + k2;
#pragma unroll
            for (int ht = 0; ht < 8; ++ht) {
                int nt2 = wave * 8 + ht;
                v16bf b2 = *(const v16bf*)(
                    w2p + ((size_t)(e * (HIDDEN / 16) + nt2) * (FFN / 32) + ktg) * 512
                        + (size_t)lane * 16);
                acc[ht] = __builtin_amdgcn_wmma_f32_16x16x32_bf16(false, a.v, false, b2,
                                                                  (short)0, acc[ht],
                                                                  false, false);
            }
        }
        __syncthreads();
    }

    // Exclusive write of this (token, rank) contribution slice.
#pragma unroll
    for (int ht = 0; ht < 8; ++ht) {
        int col = wave * 128 + ht * 16 + mlo;
#pragma unroll
        for (int r = 0; r < 8; ++r) {
            int m = r + ((lane & 16) ? 8 : 0);
            if (m < rowsLeft) {
                int trk = sm.tokrk[m];
                float* dst = (trk & 1) ? c1 : c0;
                dst[(size_t)(trk >> 1) * HIDDEN + col] = acc[ht][r];
            }
        }
    }
}

__global__ void moe_combine(const float* __restrict__ c0,
                            const float* __restrict__ c1,
                            float* __restrict__ out, size_t n)
{
    size_t i = (size_t)blockIdx.x * 256 + threadIdx.x;
    if (i < n) out[i] = c0[i] + c1[i];
}

// ---------------------------------------------------------------------------
extern "C" void kernel_launch(void* const* d_in, const int* in_sizes, int n_in,
                              void* d_out, int out_size, void* d_ws, size_t ws_size,
                              hipStream_t stream)
{
    (void)n_in; (void)out_size; (void)ws_size;
    const float* x  = (const float*)d_in[0];   // [T, H]
    const float* gw = (const float*)d_in[1];   // [H, E]
    const float* w1 = (const float*)d_in[2];   // [E, H, F]
    const float* w2 = (const float*)d_in[3];   // [E, F, H]
    const float* w3 = (const float*)d_in[4];   // [E, H, F]
    int T = in_sizes[0] / HIDDEN;

    float* out    = (float*)d_out;                      // [T*H]
    float* logits = out + (size_t)T * HIDDEN;           // [T*E]

    char* ws = (char*)d_ws;
    const size_t WE = (size_t)NEXP * HIDDEN * FFN;      // elems per weight tensor
    __bf16* w1p = (__bf16*)ws;
    __bf16* w3p = (__bf16*)(ws + WE * 2);
    __bf16* w2p = (__bf16*)(ws + WE * 4);
    char* p = ws + WE * 6;
    int*   counts = (int*)p;   p += 256;
    int*   lists  = (int*)p;   p += (size_t)NEXP * T * 4;
    int*   eA     = (int*)p;   p += (size_t)T * 4;
    int*   eB     = (int*)p;   p += (size_t)T * 4;
    float* wA     = (float*)p; p += (size_t)T * 4;
    float* wB     = (float*)p; p += (size_t)T * 4;
    float* c0     = (float*)p; p += (size_t)T * HIDDEN * 4;
    float* c1     = (float*)p;

    // Weight fp32 -> bf16 WMMA-tile packing (independent of routing).
    {
        size_t threads = (WE / 512) * 32;
        int blocks = (int)((threads + 255) / 256);
        moe_pack_bf16<<<blocks, 256, 0, stream>>>(w1, w1p, NEXP, HIDDEN, FFN);
        moe_pack_bf16<<<blocks, 256, 0, stream>>>(w3, w3p, NEXP, HIDDEN, FFN);
        moe_pack_bf16<<<blocks, 256, 0, stream>>>(w2, w2p, NEXP, FFN, HIDDEN);
    }

    moe_router<<<(T + 7) / 8, 256, 0, stream>>>(x, gw, logits, eA, eB, wA, wB, T);
    moe_zero_counts<<<1, 32, 0, stream>>>(counts);
    moe_scatter<<<(T + 255) / 256, 256, 0, stream>>>(eA, eB, counts, lists, T);

    dim3 grid(T / 16, NEXP);
    moe_expert_gemm<<<grid, 256, 0, stream>>>(x, w1p, w3p, w2p, counts, lists,
                                              wA, wB, c0, c1, T);

    size_t n = (size_t)T * HIDDEN;
    moe_combine<<<(int)((n + 255) / 256), 256, 0, stream>>>(c0, c1, out, n);
}
